// GraphAnomalyDetectionModel_83056077570930
// MI455X (gfx1250) — matrix operand order repaired and checked by reading, not compiled, not run on hardware
//
#include <hip/hip_runtime.h>
#include <hip/hip_bf16.h>
#include <math.h>

#define H_HEADS 4
#define C_CH 32
#define NEG_SLOPE 0.2f

typedef __attribute__((ext_vector_type(16))) _Float16 v16h;
typedef __attribute__((ext_vector_type(8)))  _Float16 v8h;
typedef __attribute__((ext_vector_type(8)))  float    v8f;

// ---------------------------------------------------------------------------
// Operand staging: convert f32 -> f16 with zero-padded K (Kp = ceil32(K)).
//   cvtA : A[M,K]  -> Ah[M,Kp]   (row-major, rows 16B-aligned since Kp%32==0)
//   cvtW : W[K,N]  -> Wc[N,Kp]   (transposed: one output column contiguous in K)
// ---------------------------------------------------------------------------
__global__ void cvtA_kernel(const float* __restrict__ A, _Float16* __restrict__ Ah,
                            int M, int K, int Kp)
{
    const size_t t = (size_t)blockIdx.x * blockDim.x + threadIdx.x;
    if (t >= (size_t)M * Kp) return;
    const int m = (int)(t / Kp);
    const int k = (int)(t % Kp);
    Ah[t] = (_Float16)((k < K) ? A[(size_t)m * K + k] : 0.0f);
}

__global__ void cvtW_kernel(const float* __restrict__ W, _Float16* __restrict__ Wc,
                            int K, int N, int Kp)
{
    const size_t t = (size_t)blockIdx.x * blockDim.x + threadIdx.x;
    if (t >= (size_t)N * Kp) return;
    const int n = (int)(t / Kp);
    const int k = (int)(t % Kp);
    Wc[t] = (_Float16)((k < K) ? W[(size_t)k * N + n] : 0.0f);
}

// ---------------------------------------------------------------------------
// Tall-skinny GEMM: C[M,N] = act(A[M,K] @ W[K,N] + bias[N]), f16 in / f32 acc.
// One wave = one 16x16 tile (v_wmma_f32_16x16x32_f16); 4 waves/block = 4
// consecutive M-tiles. K-loop body is branch-free: all fragment data comes
// from padded staging buffers via 16-byte loads; OOB rows are index-clamped
// (stores stay predicated). EXEC is all-ones at every WMMA.
// Fragment layouts per CDNA5 ISA 7.12.2:
//   A 16x32 f16: lanes 0-15 row=lane, v0..v3 = K 0..7, v4..v7 = K 16..23;
//                lanes 16-31 K 8..15 / 24..31.
//   B 32x16 f16: lanes 0-15 col=lane, K 0..7 / 16..23; lanes 16-31 K 8..15 / 24..31.
//   C/D 16x16 f32: vgpr i -> row i + 8*(lane/16), col = lane%16.
// ---------------------------------------------------------------------------
__global__ void gemm_wmma_f16(const _Float16* __restrict__ A,   // [M,Kp]
                              const _Float16* __restrict__ Wc,  // [N,Kp]
                              const float* __restrict__ bias,   // [N]
                              float* __restrict__ C,            // [M,N]
                              int M, int Kp, int N, int act)
{
    const int lane  = threadIdx.x & 31;
    const int wave  = threadIdx.x >> 5;
    const int mtile = blockIdx.x * 4 + wave;
    const int ntile = blockIdx.y;
    const int half  = lane >> 4;
    const int l16   = lane & 15;

    int row = mtile * 16 + l16;
    const int rowc = (row < M) ? row : (M - 1);     // clamp; stores predicated
    const int col  = ntile * 16 + l16;

    const _Float16* Ar = A  + (size_t)rowc * Kp;
    const _Float16* Wr = Wc + (size_t)col  * Kp;

    v8f acc = {};
    for (int k0 = 0; k0 < Kp; k0 += 32) {
        const int ka = k0 + half * 8;
        const v8h a_lo = *(const v8h*)(Ar + ka);
        const v8h a_hi = *(const v8h*)(Ar + ka + 16);
        const v16h a = __builtin_shufflevector(a_lo, a_hi,
                        0,1,2,3,4,5,6,7,8,9,10,11,12,13,14,15);

        const int kb = k0 + half * 8;
        const v8h b_lo = *(const v8h*)(Wr + kb);
        const v8h b_hi = *(const v8h*)(Wr + kb + 16);
        const v16h b = __builtin_shufflevector(b_lo, b_hi,
                        0,1,2,3,4,5,6,7,8,9,10,11,12,13,14,15);

        acc = __builtin_amdgcn_wmma_f32_16x16x32_f16(
                  false, a, false, b, (short)0, acc, false, false);
    }

    const float bv = bias ? bias[col] : 0.0f;
    #pragma unroll
    for (int i = 0; i < 8; ++i) {
        const int r = mtile * 16 + i + 8 * half;
        if (r < M) {
            float v = acc[i] + bv;
            if (act) v = fmaxf(v, 0.0f);
            C[(size_t)r * N + col] = v;
        }
    }
}

// ---------------------------------------------------------------------------
// Edge kernels for GATv2 segment softmax
// ---------------------------------------------------------------------------
__device__ __forceinline__ void atomicMaxF(float* addr, float val) {
    if (val >= 0.0f) atomicMax((int*)addr, __float_as_int(val));
    else             atomicMin((unsigned int*)addr, __float_as_uint(val));
}

__global__ void edge_alpha_kernel(const float* __restrict__ xl,
                                  const float* __restrict__ xr,
                                  const int* __restrict__ src,
                                  const int* __restrict__ dst,
                                  const float* __restrict__ att,        // [4,32]
                                  const float* __restrict__ edge_attr,  // [E,3] or null
                                  const float* __restrict__ We,         // [3,128] or null
                                  float* __restrict__ alpha,            // [E,4]
                                  float* __restrict__ amax,             // [Ndst,4] init -inf
                                  int E, int use_edge)
{
    const int t = blockIdx.x * blockDim.x + threadIdx.x;
    if (t >= E * H_HEADS) return;
    const int e = t >> 2;
    const int h = t & 3;
    const int s = src[e];
    const int d = dst[e];
    const float* xls = xl + (size_t)s * (H_HEADS * C_CH) + h * C_CH;
    const float* xrd = xr + (size_t)d * (H_HEADS * C_CH) + h * C_CH;
    const float* ah  = att + h * C_CH;

    float ea0 = 0.f, ea1 = 0.f, ea2 = 0.f;
    if (use_edge) {
        ea0 = edge_attr[e * 3 + 0];
        ea1 = edge_attr[e * 3 + 1];
        ea2 = edge_attr[e * 3 + 2];
    }
    float acc = 0.0f;
    #pragma unroll
    for (int c = 0; c < C_CH; ++c) {
        float x = xls[c] + xrd[c];
        if (use_edge) {
            const int cc = h * C_CH + c;
            x += ea0 * We[cc] + ea1 * We[128 + cc] + ea2 * We[256 + cc];
        }
        const float lr = (x > 0.0f) ? x : NEG_SLOPE * x;
        acc = fmaf(lr, ah[c], acc);
    }
    alpha[t] = acc;
    atomicMaxF(&amax[d * H_HEADS + h], acc);
}

__global__ void edge_expsum_kernel(const int* __restrict__ dst,
                                   float* __restrict__ alpha,
                                   const float* __restrict__ amax,
                                   float* __restrict__ denom,
                                   int E)
{
    const int t = blockIdx.x * blockDim.x + threadIdx.x;
    if (t >= E * H_HEADS) return;
    const int e = t >> 2;
    const int h = t & 3;
    const int d = dst[e];
    const float a = __expf(alpha[t] - amax[d * H_HEADS + h]);
    alpha[t] = a;
    atomicAdd(&denom[d * H_HEADS + h], a);
}

__global__ void edge_aggregate_kernel(const int* __restrict__ src,
                                      const int* __restrict__ dst,
                                      const float* __restrict__ xl,
                                      const float* __restrict__ alpha,
                                      const float* __restrict__ denom,
                                      float* __restrict__ agg,
                                      int E)
{
    const int t = blockIdx.x * blockDim.x + threadIdx.x;
    if (t >= E * 32) return;
    const int e = t >> 5;
    const int q = t & 31;          // channel quad
    const int h = q >> 3;
    const int s = src[e];
    const int d = dst[e];
    const float w = alpha[e * H_HEADS + h] / denom[d * H_HEADS + h];
    const float4 xv = *(const float4*)(xl + (size_t)s * 128 + q * 4);
    float* out = agg + (size_t)d * 128 + q * 4;
    atomicAdd(out + 0, xv.x * w);
    atomicAdd(out + 1, xv.y * w);
    atomicAdd(out + 2, xv.z * w);
    atomicAdd(out + 3, xv.w * w);
}

__global__ void fill_kernel(float* __restrict__ p, float v, size_t n)
{
    const size_t t = (size_t)blockIdx.x * blockDim.x + threadIdx.x;
    if (t < n) p[t] = v;
}

__global__ void bias_act_kernel(const float* __restrict__ in,
                                const float* __restrict__ bias,
                                float* __restrict__ out,
                                size_t n, int N, int act)
{
    const size_t t = (size_t)blockIdx.x * blockDim.x + threadIdx.x;
    if (t >= n) return;
    float v = in[t] + bias[t % (size_t)N];
    if (act) v = fmaxf(v, 0.0f);
    out[t] = v;
}

__global__ void cls_final_kernel(const float* __restrict__ h,
                                 const float* __restrict__ W2,
                                 const float* __restrict__ b2,
                                 float* __restrict__ scores,
                                 int Ncust)
{
    const int r = blockIdx.x * blockDim.x + threadIdx.x;
    if (r >= Ncust) return;
    float acc = b2[0];
    const float* hr = h + (size_t)r * 64;
    #pragma unroll
    for (int i = 0; i < 64; ++i) acc = fmaf(hr[i], W2[i], acc);
    scores[r] = 1.0f / (1.0f + __expf(-acc));
}

// ---------------------------------------------------------------------------
// Host orchestration
// ---------------------------------------------------------------------------
static inline void gemm_full(const float* A, const float* W, const float* bias,
                             float* C, int M, int K, int N, int act,
                             _Float16* Ah, _Float16* Wh, hipStream_t s)
{
    const int Kp = (K + 31) & ~31;
    {
        const size_t n = (size_t)M * Kp;
        hipLaunchKernelGGL(cvtA_kernel, dim3((unsigned)((n + 255) / 256)), dim3(256), 0, s,
                           A, Ah, M, K, Kp);
    }
    {
        const size_t n = (size_t)N * Kp;
        hipLaunchKernelGGL(cvtW_kernel, dim3((unsigned)((n + 255) / 256)), dim3(256), 0, s,
                           W, Wh, K, N, Kp);
    }
    dim3 grid((M + 63) / 64, N / 16);
    hipLaunchKernelGGL(gemm_wmma_f16, grid, dim3(128), 0, s, Ah, Wh, bias, C, M, Kp, N, act);
}

static inline void fillf(float* p, float v, size_t n, hipStream_t s)
{
    hipLaunchKernelGGL(fill_kernel, dim3((unsigned)((n + 255) / 256)), dim3(256), 0, s, p, v, n);
}

extern "C" void kernel_launch(void* const* d_in, const int* in_sizes, int n_in,
                              void* d_out, int out_size, void* d_ws, size_t ws_size,
                              hipStream_t stream)
{
    const float* customer_x = (const float*)d_in[0];
    const float* fund_x     = (const float*)d_in[1];
    const int*   edge_src   = (const int*)  d_in[2];
    const int*   edge_dst   = (const int*)  d_in[3];
    const float* edge_attr  = (const float*)d_in[4];
    const float* user_W = (const float*)d_in[5];  const float* user_b = (const float*)d_in[6];
    const float* item_W = (const float*)d_in[7];  const float* item_b = (const float*)d_in[8];
    const float* c1_Wl  = (const float*)d_in[9];  const float* c1_bl  = (const float*)d_in[10];
    const float* c1_Wr  = (const float*)d_in[11]; const float* c1_br  = (const float*)d_in[12];
    const float* c1_att = (const float*)d_in[13];
    const float* c1_We  = (const float*)d_in[14];
    const float* c1_bias= (const float*)d_in[15];
    const float* c2_Wl  = (const float*)d_in[16]; const float* c2_bl  = (const float*)d_in[17];
    const float* c2_Wr  = (const float*)d_in[18]; const float* c2_br  = (const float*)d_in[19];
    const float* c2_att = (const float*)d_in[20];
    const float* c2_bias= (const float*)d_in[21];
    const float* proj_W1= (const float*)d_in[22]; const float* proj_b1= (const float*)d_in[23];
    const float* proj_W2= (const float*)d_in[24]; const float* proj_b2= (const float*)d_in[25];
    const float* cls_W1 = (const float*)d_in[26]; const float* cls_b1 = (const float*)d_in[27];
    const float* cls_W2 = (const float*)d_in[28]; const float* cls_b2 = (const float*)d_in[29];

    const int Ncust = in_sizes[0] / 101;
    const int Nitem = in_sizes[1];
    const int E     = in_sizes[2];
    const int Nmax  = (Ncust > Nitem) ? Ncust : Nitem;

    // -------- workspace carve-out (float granules) --------
    float* ws = (float*)d_ws;
    float* user_x   = ws; ws += (size_t)Ncust * 32;
    float* item_x   = ws; ws += (size_t)Nitem * 32;
    float* xl1      = ws; ws += (size_t)Ncust * 128;
    float* xr1      = ws; ws += (size_t)Nitem * 128;
    float* xl2      = ws; ws += (size_t)Nitem * 128;
    float* xr2      = ws; ws += (size_t)Ncust * 128;
    float* alpha    = ws; ws += (size_t)E * H_HEADS;
    float* amax     = ws; ws += (size_t)Nmax * H_HEADS;
    float* denom    = ws; ws += (size_t)Nmax * H_HEADS;
    float* agg      = ws; ws += (size_t)Nmax * 128;
    float* item_out = ws; ws += (size_t)Nitem * 128;
    float* user_out = ws; ws += (size_t)Ncust * 128;
    float* ph       = ws; ws += (size_t)Ncust * 128;
    float* hbuf     = ws; ws += (size_t)Ncust * 64;
    _Float16* Ah = (_Float16*)ws; ws += (size_t)Ncust * 64;   // Ncust*128 halves
    _Float16* Wh = (_Float16*)ws; ws += 8192;                 // 128*128 halves

    float* scores = (float*)d_out;            // [Ncust]
    float* z_out  = (float*)d_out + Ncust;    // [Ncust,128]

    const int TPB = 256;
    const int gEH  = (E * H_HEADS + TPB - 1) / TPB;
    const int gE32 = (E * 32 + TPB - 1) / TPB;

    // -------- input projections --------
    gemm_full(customer_x, user_W, user_b, user_x, Ncust, 101, 32, 0, Ah, Wh, stream);
    gemm_full(fund_x,     item_W, item_b, item_x, Nitem,   1, 32, 0, Ah, Wh, stream);

    // -------- conv1: customer -> fund, with edge features --------
    gemm_full(user_x, c1_Wl, c1_bl, xl1, Ncust, 32, 128, 0, Ah, Wh, stream);
    gemm_full(item_x, c1_Wr, c1_br, xr1, Nitem, 32, 128, 0, Ah, Wh, stream);
    fillf(amax,  -INFINITY, (size_t)Nitem * H_HEADS, stream);
    fillf(denom, 0.0f,      (size_t)Nitem * H_HEADS, stream);
    fillf(agg,   0.0f,      (size_t)Nitem * 128,     stream);
    hipLaunchKernelGGL(edge_alpha_kernel, dim3(gEH), dim3(TPB), 0, stream,
                       xl1, xr1, edge_src, edge_dst, c1_att, edge_attr, c1_We,
                       alpha, amax, E, 1);
    hipLaunchKernelGGL(edge_expsum_kernel, dim3(gEH), dim3(TPB), 0, stream,
                       edge_dst, alpha, amax, denom, E);
    hipLaunchKernelGGL(edge_aggregate_kernel, dim3(gE32), dim3(TPB), 0, stream,
                       edge_src, edge_dst, xl1, alpha, denom, agg, E);
    hipLaunchKernelGGL(bias_act_kernel, dim3((unsigned)(((size_t)Nitem * 128 + TPB - 1) / TPB)),
                       dim3(TPB), 0, stream, agg, c1_bias, item_out,
                       (size_t)Nitem * 128, 128, 1);  // relu

    // -------- conv2: fund -> customer (flipped edges, no edge attr) --------
    gemm_full(item_out, c2_Wl, c2_bl, xl2, Nitem, 128, 128, 0, Ah, Wh, stream);
    gemm_full(user_x,   c2_Wr, c2_br, xr2, Ncust,  32, 128, 0, Ah, Wh, stream);
    fillf(amax,  -INFINITY, (size_t)Ncust * H_HEADS, stream);
    fillf(denom, 0.0f,      (size_t)Ncust * H_HEADS, stream);
    fillf(agg,   0.0f,      (size_t)Ncust * 128,     stream);
    hipLaunchKernelGGL(edge_alpha_kernel, dim3(gEH), dim3(TPB), 0, stream,
                       xl2, xr2, edge_dst, edge_src, c2_att, (const float*)nullptr,
                       (const float*)nullptr, alpha, amax, E, 0);
    hipLaunchKernelGGL(edge_expsum_kernel, dim3(gEH), dim3(TPB), 0, stream,
                       edge_src, alpha, amax, denom, E);
    hipLaunchKernelGGL(edge_aggregate_kernel, dim3(gE32), dim3(TPB), 0, stream,
                       edge_dst, edge_src, xl2, alpha, denom, agg, E);
    hipLaunchKernelGGL(bias_act_kernel, dim3((unsigned)(((size_t)Ncust * 128 + TPB - 1) / TPB)),
                       dim3(TPB), 0, stream, agg, c2_bias, user_out,
                       (size_t)Ncust * 128, 128, 0);  // no relu

    // -------- projection head: z = relu(u@W1+b1)@W2+b2 --------
    gemm_full(user_out, proj_W1, proj_b1, ph,    Ncust, 128, 128, 1, Ah, Wh, stream);
    gemm_full(ph,       proj_W2, proj_b2, z_out, Ncust, 128, 128, 0, Ah, Wh, stream);

    // -------- classifier: sigmoid(relu(u@W1+b1)@W2+b2) --------
    gemm_full(user_out, cls_W1, cls_b1, hbuf, Ncust, 128, 64, 1, Ah, Wh, stream);
    hipLaunchKernelGGL(cls_final_kernel, dim3((Ncust + TPB - 1) / TPB), dim3(TPB), 0, stream,
                       hbuf, cls_W2, cls_b2, scores, Ncust);
}
